// TransformerBlock_12592844112289
// MI455X (gfx1250) — matrix operand order repaired and checked by reading, not compiled, not run on hardware
//
#include <hip/hip_runtime.h>

// ============================================================================
// Transformer block (pre-LN attn + FFN) for MI455X / gfx1250 (CDNA5, wave32).
// Matrix math in bf16 via V_WMMA_F32_16X16X32_BF16 (f32 accumulate); LN,
// softmax, GELU, residuals in f32. Flash-style causal attention.
// - Weights pre-transposed to [N][K] bf16 (tile fills = pure b128 copies)
// - V projection written directly in [B][H][HD][T] layout by the GEMM
//   epilogue, so attention never transposes through LDS
// - Softmax row reductions via DPP16 (no LDS traffic)
// - GLOBAL_LOAD_ASYNC_TO_LDS_B128 tile fills (ASYNCcnt) when the builtin is
//   available; falls back to register-staged copies otherwise
// ============================================================================

typedef __attribute__((ext_vector_type(16))) __bf16 v16bf;
typedef __attribute__((ext_vector_type(8)))  float  v8f;
typedef __attribute__((ext_vector_type(4)))  int    v4i;

#define DMODEL 1024
#define FFDIM  4096
#define NHEAD  16
#define HDIM   64
#define TDIM   2048

#define AS1 __attribute__((address_space(1)))
#define AS3 __attribute__((address_space(3)))

#if __has_builtin(__builtin_amdgcn_global_load_async_to_lds_b128)
#define HAVE_ASYNC_LDS 1
__device__ inline void async_g2l_b128(__bf16* lds_dst, const __bf16* gsrc) {
  v4i* gp = (v4i*)gsrc;  // drop const; still generic
  __builtin_amdgcn_global_load_async_to_lds_b128(
      (AS1 v4i*)gp, (AS3 v4i*)lds_dst, 0, 0);
}
__device__ inline void async_wait0() {
#if __has_builtin(__builtin_amdgcn_s_wait_asynccnt)
  __builtin_amdgcn_s_wait_asynccnt(0);
#else
  asm volatile("s_wait_asynccnt 0x0" ::: "memory");
#endif
}
#else
#define HAVE_ASYNC_LDS 0
#endif

__device__ inline __bf16 f2bf(float f) {
  unsigned u = __builtin_bit_cast(unsigned, f);
  unsigned r = u + 0x7FFFu + ((u >> 16) & 1u);   // round-to-nearest-even
  return __builtin_bit_cast(__bf16, (unsigned short)(r >> 16));
}

__device__ inline v8f vzero8() {
  v8f z;
#pragma unroll
  for (int i = 0; i < 8; ++i) z[i] = 0.0f;
  return z;
}

// ---------------------------------------------------------------------------
// 16-lane group reductions (rows of the WMMA C layout) via DPP16.
// ---------------------------------------------------------------------------
#if __has_builtin(__builtin_amdgcn_update_dpp)
template <int CTRL>
__device__ inline float dpp_perm(float v) {
  return __builtin_bit_cast(
      float, __builtin_amdgcn_update_dpp(0, __builtin_bit_cast(int, v), CTRL,
                                         0xF, 0xF, true));
}
__device__ inline float redmax16(float v) {
  v = fmaxf(v, dpp_perm<0xB1>(v));    // quad_perm(1,0,3,2)
  v = fmaxf(v, dpp_perm<0x4E>(v));    // quad_perm(2,3,0,1)
  v = fmaxf(v, dpp_perm<0x124>(v));   // row_ror:4
  v = fmaxf(v, dpp_perm<0x128>(v));   // row_ror:8
  return v;
}
__device__ inline float redsum16(float v) {
  v += dpp_perm<0xB1>(v);
  v += dpp_perm<0x4E>(v);
  v += dpp_perm<0x124>(v);
  v += dpp_perm<0x128>(v);
  return v;
}
#else
__device__ inline float redmax16(float v) {
  v = fmaxf(v, __shfl_xor(v, 1));
  v = fmaxf(v, __shfl_xor(v, 2));
  v = fmaxf(v, __shfl_xor(v, 4));
  v = fmaxf(v, __shfl_xor(v, 8));
  return v;
}
__device__ inline float redsum16(float v) {
  v += __shfl_xor(v, 1);
  v += __shfl_xor(v, 2);
  v += __shfl_xor(v, 4);
  v += __shfl_xor(v, 8);
  return v;
}
#endif

// ---------------------------------------------------------------------------
// WMMA fragment loaders per CDNA5 ISA §7.12.2 (wave32).
// A (16x32 bf16): lane L holds row M=L&15;
//   VGPR v, elem e -> K = ((v>>2)<<4) + ((v&3)<<1) + ((L>>4)<<3) + e
// B (32x16 bf16) sourced from an LDS tile stored [N][K]:
//   lane L holds col N=L&15; K = ((v>>2)<<4) + ((v&3)<<1) + ((L>>4)<<3) + e
// C/D (16x16 f32): VGPR g -> row M = g + ((L>>4)<<3), col N = L&15
// ---------------------------------------------------------------------------
__device__ inline v16bf ldfrag_a(const __bf16* __restrict__ tile, int ld, int lane) {
  const __bf16* p = tile + (lane & 15) * ld;
  const int kh = (lane >> 4) << 3;
  v16bf a;
#pragma unroll
  for (int v = 0; v < 8; ++v) {
    int k = ((v >> 2) << 4) + ((v & 3) << 1) + kh;
    a[2 * v]     = p[k];
    a[2 * v + 1] = p[k + 1];
  }
  return a;
}

__device__ inline v16bf ldfrag_b(const __bf16* __restrict__ tileNK, int ld, int lane) {
  const __bf16* p = tileNK + (lane & 15) * ld + ((lane >> 4) << 3);
  v16bf b;
#pragma unroll
  for (int v = 0; v < 8; ++v) {
    int k = ((v >> 2) << 4) + ((v & 3) << 1);
    b[2 * v]     = p[k];
    b[2 * v + 1] = p[k + 1];
  }
  return b;
}

__device__ inline v8f wmma_bf16(v16bf a, v16bf b, v8f c) {
  // (neg_a, A, neg_b, B, c_mod, C, reuse_a, reuse_b)
  return __builtin_amdgcn_wmma_f32_16x16x32_bf16(false, a, false, b, (short)0, c,
                                                 false, false);
}

// ---------------------------------------------------------------------------
// Fused fp32->bf16 convert + transpose (weights -> [N][K]), 32x32 LDS tiles.
// ---------------------------------------------------------------------------
__global__ __launch_bounds__(256) void transpose_cvt_f32_bf16(
    const float* __restrict__ in, __bf16* __restrict__ out, int R, int C) {
  __shared__ float tile[32][33];
  const int c0 = blockIdx.x * 32, r0 = blockIdx.y * 32;
  const int tx = threadIdx.x & 31, ty = threadIdx.x >> 5;
#pragma unroll
  for (int i = 0; i < 32; i += 8)
    tile[ty + i][tx] = in[(size_t)(r0 + ty + i) * C + c0 + tx];
  __syncthreads();
#pragma unroll
  for (int i = 0; i < 32; i += 8)
    out[(size_t)(c0 + ty + i) * R + r0 + tx] = f2bf(tile[tx][ty + i]);
}

// ---------------------------------------------------------------------------
// LayerNorm over D=1024: one block (8 waves) per row, bf16 output
// ---------------------------------------------------------------------------
__global__ __launch_bounds__(256) void ln_bf16_kernel(
    const float* __restrict__ X, const float* __restrict__ scale,
    const float* __restrict__ shift, __bf16* __restrict__ Out) {
  __shared__ float red[16];
  const int row = blockIdx.x, t = threadIdx.x;
  const float* xr = X + (size_t)row * DMODEL;
  float xv[4];
  float s = 0.0f, s2 = 0.0f;
#pragma unroll
  for (int i = 0; i < 4; ++i) {
    xv[i] = xr[t + i * 256];
    s += xv[i];
    s2 += xv[i] * xv[i];
  }
#pragma unroll
  for (int off = 1; off < 32; off <<= 1) {
    s += __shfl_xor(s, off);
    s2 += __shfl_xor(s2, off);
  }
  const int lane = t & 31, wid = t >> 5;
  if (lane == 0) { red[wid] = s; red[8 + wid] = s2; }
  __syncthreads();
  float ts = 0.0f, ts2 = 0.0f;
#pragma unroll
  for (int w = 0; w < 8; ++w) { ts += red[w]; ts2 += red[8 + w]; }
  const float mean = ts * (1.0f / DMODEL);
  const float var  = ts2 * (1.0f / DMODEL) - mean * mean;
  const float rstd = rsqrtf(var + 1e-5f);
#pragma unroll
  for (int i = 0; i < 4; ++i) {
    int c = t + i * 256;
    Out[(size_t)row * DMODEL + c] = f2bf(scale[c] * ((xv[i] - mean) * rstd) + shift[c]);
  }
}

// ---------------------------------------------------------------------------
// Tiled bf16 WMMA GEMM: C(MxN) = A(MxK) * Bt^T, A row-major, Bt pre-
// transposed [N][K]. Block 256 (8 waves), BM=128, BN=256, BK=32.
// Waves 2(M)x4(N); each wave owns 64x64 = 4x4 subtiles -> 16 WMMAs per
// k-step from 4 A-frags + 4 B-frags (2 WMMAs per fragment load).
// ---------------------------------------------------------------------------
enum { EPI_BF16 = 0, EPI_BIAS_RES_F32 = 1, EPI_GELU_BF16 = 2, EPI_BF16T = 3 };

template <int EPI>
__global__ __launch_bounds__(256) void gemm_bf16_wmma(
    const __bf16* __restrict__ A, const __bf16* __restrict__ Bt,
    void* __restrict__ Out, const float* __restrict__ bias,
    const float* __restrict__ resid, int M, int N, int K) {
  constexpr int BM = 128, BN = 256, BK = 32, LDA = 40, LDB = 40;
  __shared__ __bf16 As[BM * LDA];   // [m][k]
  __shared__ __bf16 Bs[BN * LDB];   // [n][k]
  const int t = threadIdx.x, lane = t & 31, wid = t >> 5;
  const int wm = (wid & 1) * 64, wn = (wid >> 1) * 64;
  const int m0 = blockIdx.y * BM, n0 = blockIdx.x * BN;

  v8f acc[4][4];
#pragma unroll
  for (int i = 0; i < 4; ++i)
#pragma unroll
    for (int j = 0; j < 4; ++j) acc[i][j] = vzero8();

  const int arow = t >> 1, ak = (t & 1) * 16;  // 2 threads per A row
  const __bf16* gA = A + (size_t)(m0 + arow) * K + ak;
  const __bf16* gB = Bt + (size_t)(n0 + t) * K;  // 1 thread per B row

  for (int k0 = 0; k0 < K; k0 += BK) {
#if HAVE_ASYNC_LDS
    __syncthreads();  // previous iteration's LDS reads done
    async_g2l_b128(&As[arow * LDA + ak],     gA + k0);
    async_g2l_b128(&As[arow * LDA + ak + 8], gA + k0 + 8);
    async_g2l_b128(&Bs[t * LDB],      gB + k0);
    async_g2l_b128(&Bs[t * LDB + 8],  gB + k0 + 8);
    async_g2l_b128(&Bs[t * LDB + 16], gB + k0 + 16);
    async_g2l_b128(&Bs[t * LDB + 24], gB + k0 + 24);
    async_wait0();
    __syncthreads();
#else
    uint4 a0 = *(const uint4*)(gA + k0);
    uint4 a1 = *(const uint4*)(gA + k0 + 8);
    uint4 b0 = *(const uint4*)(gB + k0);
    uint4 b1 = *(const uint4*)(gB + k0 + 8);
    uint4 b2 = *(const uint4*)(gB + k0 + 16);
    uint4 b3 = *(const uint4*)(gB + k0 + 24);
    if (k0 + BK < K) {
      __builtin_prefetch(gA + k0 + BK, 0, 1);  // global_prefetch_b8
      __builtin_prefetch(gB + k0 + BK, 0, 1);
    }
    __syncthreads();  // previous iteration's LDS reads done
    *(uint4*)&As[arow * LDA + ak]     = a0;
    *(uint4*)&As[arow * LDA + ak + 8] = a1;
    *(uint4*)&Bs[t * LDB]      = b0;
    *(uint4*)&Bs[t * LDB + 8]  = b1;
    *(uint4*)&Bs[t * LDB + 16] = b2;
    *(uint4*)&Bs[t * LDB + 24] = b3;
    __syncthreads();
#endif

    v16bf af[4], bfr[4];
#pragma unroll
    for (int i = 0; i < 4; ++i) af[i] = ldfrag_a(&As[(wm + i * 16) * LDA], LDA, lane);
#pragma unroll
    for (int j = 0; j < 4; ++j) bfr[j] = ldfrag_b(&Bs[(wn + j * 16) * LDB], LDB, lane);
#pragma unroll
    for (int i = 0; i < 4; ++i)
#pragma unroll
      for (int j = 0; j < 4; ++j) acc[i][j] = wmma_bf16(af[i], bfr[j], acc[i][j]);
  }

  const int nloc = lane & 15, mb = (lane >> 4) << 3;
#pragma unroll
  for (int i = 0; i < 4; ++i)
#pragma unroll
    for (int j = 0; j < 4; ++j) {
      const int gn = n0 + wn + j * 16 + nloc;
#pragma unroll
      for (int g = 0; g < 8; ++g) {
        const int gm = m0 + wm + i * 16 + mb + g;
        float v = acc[i][j][g];
        if (EPI == EPI_BF16) {
          ((__bf16*)Out)[(size_t)gm * N + gn] = f2bf(v);
        } else if (EPI == EPI_BIAS_RES_F32) {
          const size_t o = (size_t)gm * N + gn;
          ((float*)Out)[o] = v + bias[gn] + resid[o];
        } else if (EPI == EPI_GELU_BF16) {  // bias + tanh-GELU -> bf16
          v += bias[gn];
          float inner = 0.7978845608028654f * (v + 0.044715f * v * v * v);
          ((__bf16*)Out)[(size_t)gm * N + gn] = f2bf(0.5f * v * (1.0f + tanhf(inner)));
        } else {  // EPI_BF16T: V projection -> [B][H][HD][T]
          const int h = gn >> 6, d = gn & (HDIM - 1);
          const int bb = gm / TDIM, tok = gm % TDIM;  // consecutive over g
          ((__bf16*)Out)[((((size_t)bb * NHEAD) + h) * HDIM + d) * TDIM + tok] =
              f2bf(v);
        }
      }
    }
}

// ---------------------------------------------------------------------------
// Flash attention (causal). One block per (b, h, 128-row Q tile); 8 waves,
// each wave owns 16 q-rows. Key tiles of 32 stream through LDS.
// V arrives pre-transposed ([B][H][HD][T]) so every tile fill is b128 copies.
// ---------------------------------------------------------------------------
__global__ __launch_bounds__(256) void attn_flash_wmma(
    const __bf16* __restrict__ Qb, const __bf16* __restrict__ Kb,
    const __bf16* __restrict__ Vtg, __bf16* __restrict__ Ctx, int T) {
  __shared__ __bf16 Qs[128 * 72];  // [q][d]
  __shared__ __bf16 Ks[32 * 72];   // [key][d]  == B-tile [N=key][K=d] for S
  __shared__ __bf16 Vt[64 * 40];   // [d][key]  == B-tile [N=d][K=key] for O
  __shared__ __bf16 Ps[128 * 40];  // P staging (C-layout -> A-layout reshuffle)
  const int t = threadIdx.x, lane = t & 31, wid = t >> 5;
  const int b = blockIdx.z, h = blockIdx.y, q0 = blockIdx.x * 128;
  const size_t base  = ((size_t)b * T) * DMODEL + (size_t)h * HDIM;
  const size_t vbase = (((size_t)b * NHEAD) + h) * HDIM * (size_t)T;

  // load 128x64 Q tile
#pragma unroll
  for (int i = 0; i < 4; ++i) {
    int chunk = t + i * 256;
    int r = chunk >> 3, c = (chunk & 7) * 8;
#if HAVE_ASYNC_LDS
    async_g2l_b128(&Qs[r * 72 + c], Qb + base + (size_t)(q0 + r) * DMODEL + c);
#else
    *(uint4*)&Qs[r * 72 + c] =
        *(const uint4*)(Qb + base + (size_t)(q0 + r) * DMODEL + c);
#endif
  }
  const int qrow0 = wid * 16, mb = (lane >> 4) << 3, nloc = lane & 15;
  float mrow[8], lrow[8];
  v8f o[4];
#pragma unroll
  for (int g = 0; g < 8; ++g) { mrow[g] = -1e30f; lrow[g] = 0.0f; }
#pragma unroll
  for (int d = 0; d < 4; ++d) o[d] = vzero8();
#if HAVE_ASYNC_LDS
  async_wait0();
#endif
  __syncthreads();
  const v16bf aq0 = ldfrag_a(&Qs[qrow0 * 72], 72, lane);       // d = 0..31
  const v16bf aq1 = ldfrag_a(&Qs[qrow0 * 72 + 32], 72, lane);  // d = 32..63

  const int kr = t >> 3, kc = (t & 7) * 8;   // K tile: 32x64
  const int vd = t >> 2, vc = (t & 3) * 8;   // V tile: 64x32 (already [d][tok])
  const int ntile = (q0 + 128) / 32;         // causal upper bound
  for (int kt = 0; kt < ntile; ++kt) {
    const int k0 = kt * 32;
#if HAVE_ASYNC_LDS
    __syncthreads();  // previous iteration's LDS reads done
    async_g2l_b128(&Ks[kr * 72 + kc], Kb + base + (size_t)(k0 + kr) * DMODEL + kc);
    async_g2l_b128(&Vt[vd * 40 + vc], Vtg + vbase + (size_t)vd * T + k0 + vc);
    async_wait0();
    __syncthreads();
#else
    uint4 kv = *(const uint4*)(Kb + base + (size_t)(k0 + kr) * DMODEL + kc);
    uint4 vv = *(const uint4*)(Vtg + vbase + (size_t)vd * T + k0 + vc);
    __syncthreads();
    *(uint4*)&Ks[kr * 72 + kc] = kv;
    *(uint4*)&Vt[vd * 40 + vc] = vv;
    __syncthreads();
#endif

    // S(16x32) = Q(16x64) * K^T, two 16-key subtiles, K-dim split 32+32
    v8f s[2];
#pragma unroll
    for (int nt = 0; nt < 2; ++nt) {
      v8f sv = vzero8();
      sv = wmma_bf16(aq0, ldfrag_b(&Ks[nt * 16 * 72], 72, lane), sv);
      sv = wmma_bf16(aq1, ldfrag_b(&Ks[nt * 16 * 72 + 32], 72, lane), sv);
      s[nt] = sv;
    }
#pragma unroll
    for (int nt = 0; nt < 2; ++nt)
#pragma unroll
      for (int g = 0; g < 8; ++g) {
        int qi = q0 + qrow0 + mb + g;
        int ki = k0 + nt * 16 + nloc;
        float v = s[nt][g] * 0.125f;  // 1/sqrt(HD)
        s[nt][g] = (ki > qi) ? -1e30f : v;
      }
    // online softmax update (per row; rows live across 16-lane groups)
    float mnew[8], sc[8], p0[8], p1[8];
#pragma unroll
    for (int g = 0; g < 8; ++g) {
      float rm = redmax16(fmaxf(s[0][g], s[1][g]));
      mnew[g] = fmaxf(mrow[g], rm);
      sc[g]   = __expf(mrow[g] - mnew[g]);
      mrow[g] = mnew[g];
      p0[g] = __expf(s[0][g] - mnew[g]);
      p1[g] = __expf(s[1][g] - mnew[g]);
      lrow[g] = lrow[g] * sc[g] + redsum16(p0[g] + p1[g]);
    }
#pragma unroll
    for (int d = 0; d < 4; ++d)
#pragma unroll
      for (int g = 0; g < 8; ++g) o[d][g] *= sc[g];
    // stage P through LDS (per-wave region; in-wave DS ordering via dscnt)
#pragma unroll
    for (int g = 0; g < 8; ++g) {
      Ps[(qrow0 + mb + g) * 40 + nloc]      = f2bf(p0[g]);
      Ps[(qrow0 + mb + g) * 40 + 16 + nloc] = f2bf(p1[g]);
    }
    const v16bf ap = ldfrag_a(&Ps[qrow0 * 40], 40, lane);  // P as A (16x32)
#pragma unroll
    for (int d = 0; d < 4; ++d)
      o[d] = wmma_bf16(ap, ldfrag_b(&Vt[d * 16 * 40], 40, lane), o[d]);
  }
#pragma unroll
  for (int d = 0; d < 4; ++d)
#pragma unroll
    for (int g = 0; g < 8; ++g) {
      int qi = q0 + qrow0 + mb + g;
      Ctx[base + (size_t)qi * DMODEL + d * 16 + nloc] = f2bf(o[d][g] / lrow[g]);
    }
}

// ---------------------------------------------------------------------------
// Launcher
// ---------------------------------------------------------------------------
extern "C" void kernel_launch(void* const* d_in, const int* in_sizes, int n_in,
                              void* d_out, int out_size, void* d_ws, size_t ws_size,
                              hipStream_t stream) {
  (void)in_sizes; (void)n_in; (void)out_size; (void)ws_size;
  const float* x    = (const float*)d_in[0];
  const float* Wq   = (const float*)d_in[1];
  const float* Wk   = (const float*)d_in[2];
  const float* Wv   = (const float*)d_in[3];
  const float* Wo   = (const float*)d_in[4];
  const float* bo   = (const float*)d_in[5];
  const float* W1   = (const float*)d_in[6];
  const float* b1   = (const float*)d_in[7];
  const float* W2   = (const float*)d_in[8];
  const float* b2   = (const float*)d_in[9];
  const float* ln1s = (const float*)d_in[10];
  const float* ln1b = (const float*)d_in[11];
  const float* ln2s = (const float*)d_in[12];
  const float* ln2b = (const float*)d_in[13];

  const int Bb = 4, T = TDIM, D = DMODEL, FF = FFDIM;
  const int M = Bb * T;  // 8192

  char* p = (char*)d_ws;
  auto alloc = [&](size_t bytes) {
    char* r = p;
    p += (bytes + 255) & ~(size_t)255;
    return r;
  };
  // Pre-transposed bf16 weights: [N][K]
  __bf16* wqb = (__bf16*)alloc((size_t)D * D * 2);
  __bf16* wkb = (__bf16*)alloc((size_t)D * D * 2);
  __bf16* wvb = (__bf16*)alloc((size_t)D * D * 2);
  __bf16* wob = (__bf16*)alloc((size_t)D * D * 2);
  __bf16* w1b = (__bf16*)alloc((size_t)D * FF * 2);
  __bf16* w2b = (__bf16*)alloc((size_t)FF * D * 2);
  __bf16* hb  = (__bf16*)alloc((size_t)M * D * 2);   // LN output (reused)
  __bf16* qb  = (__bf16*)alloc((size_t)M * D * 2);
  __bf16* kb  = (__bf16*)alloc((size_t)M * D * 2);
  __bf16* vt  = (__bf16*)alloc((size_t)M * D * 2);   // V in [B][H][HD][T]
  __bf16* cb  = (__bf16*)alloc((size_t)M * D * 2);   // attention context
  float*  x1  = (float*)alloc((size_t)M * D * 4);    // x + attn residual
  __bf16* gb  = (__bf16*)alloc((size_t)M * FF * 2);  // gelu(h@W1+b1)

  dim3 blk(256);
  // convert + transpose weights once (amortized over 64 M-tiles of reuse)
  dim3 gtD(D / 32, D / 32);
  transpose_cvt_f32_bf16<<<gtD, blk, 0, stream>>>(Wq, wqb, D, D);
  transpose_cvt_f32_bf16<<<gtD, blk, 0, stream>>>(Wk, wkb, D, D);
  transpose_cvt_f32_bf16<<<gtD, blk, 0, stream>>>(Wv, wvb, D, D);
  transpose_cvt_f32_bf16<<<gtD, blk, 0, stream>>>(Wo, wob, D, D);
  dim3 gt1(FF / 32, D / 32);
  transpose_cvt_f32_bf16<<<gt1, blk, 0, stream>>>(W1, w1b, D, FF);
  dim3 gt2(D / 32, FF / 32);
  transpose_cvt_f32_bf16<<<gt2, blk, 0, stream>>>(W2, w2b, FF, D);

  ln_bf16_kernel<<<M, blk, 0, stream>>>(x, ln1s, ln1b, hb);

  dim3 g1(D / 256, M / 128);  // (4, 64)
  gemm_bf16_wmma<EPI_BF16><<<g1, blk, 0, stream>>>(hb, wqb, qb, nullptr, nullptr, M, D, D);
  gemm_bf16_wmma<EPI_BF16><<<g1, blk, 0, stream>>>(hb, wkb, kb, nullptr, nullptr, M, D, D);
  gemm_bf16_wmma<EPI_BF16T><<<g1, blk, 0, stream>>>(hb, wvb, vt, nullptr, nullptr, M, D, D);

  dim3 ga(T / 128, NHEAD, Bb);
  attn_flash_wmma<<<ga, blk, 0, stream>>>(qb, kb, vt, cb, T);

  gemm_bf16_wmma<EPI_BIAS_RES_F32><<<g1, blk, 0, stream>>>(cb, wob, x1, bo, x, M, D, D);

  ln_bf16_kernel<<<M, blk, 0, stream>>>(x1, ln2s, ln2b, hb);

  dim3 g2(FF / 256, M / 128);  // (16, 64)
  gemm_bf16_wmma<EPI_GELU_BF16><<<g2, blk, 0, stream>>>(hb, w1b, gb, b1, nullptr, M, FF, D);

  gemm_bf16_wmma<EPI_BIAS_RES_F32><<<g1, blk, 0, stream>>>(gb, w2b, (float*)d_out, b2, x1, M, D, FF);
}
